// TLGNN_23201413333310
// MI455X (gfx1250) — matrix operand order repaired and checked by reading, not compiled, not run on hardware
//
#include <hip/hip_runtime.h>
#include <stdint.h>

// ---------------------------------------------------------------------------
// TLGNN on MI455X (gfx1250): bf16 WMMA GEMMs + f32 atomics for sparse agg.
// GEMM: Y = X(f32,NxK) * W(KxM) + b via v_wmma_f32_16x16x32_bf16.
// Block tile 64x128, 8 wave32s (4 along M x 2 along N), each wave computes a
// 16x64 strip = 4 WMMA accumulators per K-stage. A sched_barrier between the
// fragment loads and the WMMA burst forces all B fragments into distinct
// registers so the 4 matrix ops issue back-to-back behind one s_wait_dscnt.
// ---------------------------------------------------------------------------

#define N_NODES 50000
#define N_EDGES 800000
#define DIM_IN  256
#define HID     512
#define DOUT    16
#define N_G     128
#define BN_EPS  1e-5f

typedef __attribute__((ext_vector_type(16))) __bf16 v16bf;
typedef __attribute__((ext_vector_type(8)))  float  v8f;

__device__ __forceinline__ unsigned short f2bf(float f) {
  unsigned int u = __float_as_uint(f);
  u += 0x7FFFu + ((u >> 16) & 1u);   // round-to-nearest-even
  return (unsigned short)(u >> 16);
}

struct FragBF { union { v16bf v; uint4 q[2]; }; };

// ------------------------------- GEMM --------------------------------------
#define BM 64
#define BN 128
#define BK 32

__global__ __launch_bounds__(256) void gemm_bf16_k(
    float* __restrict__ Y, const float* __restrict__ X,
    const unsigned short* __restrict__ WT, const float* __restrict__ bias,
    int n, int K, int M) {
  __shared__ __align__(16) unsigned short As[BM][BK];   // 4 KB
  __shared__ __align__(16) unsigned short Bs[BN][BK];   // 8 KB
  const int tid  = threadIdx.x;
  const int lane = tid & 31;
  const int w    = tid >> 5;       // wave 0..7
  const int wm   = w & 3;          // 4 waves along M (4*16 = 64 rows)
  const int wn   = w >> 2;         // 2 waves along N (2*64 = 128 cols)
  const int hl   = lane >> 4;      // lane half
  const int l16  = lane & 15;
  const int m0   = blockIdx.x * BM;
  const int n0   = blockIdx.y * BN;

  v8f acc[4];
  #pragma unroll
  for (int f = 0; f < 4; ++f) acc[f] = (v8f){};

  // A staging: 64x32 tile, 8 bf16 per thread
  const int sr  = tid >> 2;        // 0..63
  const int sc  = (tid & 3) * 8;   // 0,8,16,24
  // B staging: 128x32 tile, 16 bf16 per thread
  const int sr2 = tid >> 1;        // 0..127
  const int sc2 = (tid & 1) * 16;  // 0,16

  for (int k0 = 0; k0 < K; k0 += BK) {
    // stage A tile (f32 -> bf16)
    {
      unsigned short tmp[8];
      int row = m0 + sr;
      if (row < n) {
        const float* src = X + (size_t)row * K + k0 + sc;
        float4 f0 = *(const float4*)(src);
        float4 f1 = *(const float4*)(src + 4);
        tmp[0]=f2bf(f0.x); tmp[1]=f2bf(f0.y); tmp[2]=f2bf(f0.z); tmp[3]=f2bf(f0.w);
        tmp[4]=f2bf(f1.x); tmp[5]=f2bf(f1.y); tmp[6]=f2bf(f1.z); tmp[7]=f2bf(f1.w);
      } else {
        #pragma unroll
        for (int i = 0; i < 8; ++i) tmp[i] = 0;
      }
      *(uint4*)&As[sr][sc] = *(const uint4*)tmp;
      // stage B tile (already bf16, WT is [M][K] row-major)
      const uint4* bsrc = (const uint4*)(WT + (size_t)(n0 + sr2) * K + k0 + sc2);
      *(uint4*)&Bs[sr2][sc2 + 0] = bsrc[0];
      *(uint4*)&Bs[sr2][sc2 + 8] = bsrc[1];
    }
    if (k0 + BK < K) {
      __builtin_prefetch(WT + (size_t)(n0 + sr2) * K + (k0 + BK) + sc2, 0, 3);
      if (m0 + sr < n)
        __builtin_prefetch(X + (size_t)(m0 + sr) * K + (k0 + BK) + sc, 0, 3);
    }
    __syncthreads();

    // Load ALL fragments first, fence the scheduler, then the WMMA burst.
    // A frag: row = wm*16+l16; K run = contiguous hl*16..hl*16+15
    FragBF fa, fb0, fb1, fb2, fb3;
    const uint4* ap = (const uint4*)&As[wm * 16 + l16][0];
    fa.q[0] = ap[hl * 2];
    fa.q[1] = ap[hl * 2 + 1];
    // B frags: col = wn*64 + f*16 + l16; K run = contiguous hl*16..hl*16+15
    const uint4* bp0 = (const uint4*)&Bs[wn * 64 +  0 + l16][0];
    const uint4* bp1 = (const uint4*)&Bs[wn * 64 + 16 + l16][0];
    const uint4* bp2 = (const uint4*)&Bs[wn * 64 + 32 + l16][0];
    const uint4* bp3 = (const uint4*)&Bs[wn * 64 + 48 + l16][0];
    fb0.q[0] = bp0[hl * 2]; fb0.q[1] = bp0[hl * 2 + 1];
    fb1.q[0] = bp1[hl * 2]; fb1.q[1] = bp1[hl * 2 + 1];
    fb2.q[0] = bp2[hl * 2]; fb2.q[1] = bp2[hl * 2 + 1];
    fb3.q[0] = bp3[hl * 2]; fb3.q[1] = bp3[hl * 2 + 1];

    // Do not let the scheduler sink the loads into the WMMA sequence.
    __builtin_amdgcn_sched_barrier(0);

    acc[0] = __builtin_amdgcn_wmma_f32_16x16x32_bf16(false, fa.v, false, fb0.v,
                                                     (short)0, acc[0], false, false);
    acc[1] = __builtin_amdgcn_wmma_f32_16x16x32_bf16(false, fa.v, false, fb1.v,
                                                     (short)0, acc[1], false, false);
    acc[2] = __builtin_amdgcn_wmma_f32_16x16x32_bf16(false, fa.v, false, fb2.v,
                                                     (short)0, acc[2], false, false);
    acc[3] = __builtin_amdgcn_wmma_f32_16x16x32_bf16(false, fa.v, false, fb3.v,
                                                     (short)0, acc[3], false, false);
    __syncthreads();
  }

  // epilogue: C/D layout — VGPR r: lanes 0-15 -> M=r, lanes 16-31 -> M=8+r
  #pragma unroll
  for (int f = 0; f < 4; ++f) {
    int col = n0 + wn * 64 + f * 16 + l16;
    float bv = bias[col];
    #pragma unroll
    for (int r = 0; r < 8; ++r) {
      int row = m0 + wm * 16 + hl * 8 + r;
      if (row < n) Y[(size_t)row * M + col] = acc[f][r] + bv;
    }
  }
}

// --------------------------- support kernels -------------------------------

__global__ void convert_wt_k(unsigned short* __restrict__ WT,
                             const float* __restrict__ W, int K, int M) {
  int t = blockIdx.x * blockDim.x + threadIdx.x;
  if (t >= K * M) return;
  int m = t / K, k = t % K;
  WT[t] = f2bf(W[(size_t)k * M + m]);   // WT[m][k] = W[k][m]
}

__global__ void zero_k(float* __restrict__ p, int total) {
  int t = blockIdx.x * blockDim.x + threadIdx.x;
  if (t < total) p[t] = 0.0f;
}

__global__ void scale_init_k(float* __restrict__ out, const float* __restrict__ src,
                             const float* __restrict__ eps, int l, unsigned total) {
  unsigned t = blockIdx.x * blockDim.x + threadIdx.x;
  if (t >= total) return;
  out[t] = (1.0f + eps[l]) * src[t];
}

__global__ void edge_agg4_k(const int* __restrict__ src, const int* __restrict__ dst,
                            const float* __restrict__ hsrc, float* __restrict__ pooled,
                            int D) {
  unsigned D4 = (unsigned)D >> 2;
  unsigned total = (unsigned)N_EDGES * D4;
  unsigned t = blockIdx.x * blockDim.x + threadIdx.x;
  if (t >= total) return;
  unsigned e = t / D4;
  unsigned c = (t % D4) << 2;
  int s = src[e], d = dst[e];
  float4 v = *(const float4*)(hsrc + (size_t)s * D + c);
  float* p = pooled + (size_t)d * D + c;
  atomicAdd(p + 0, v.x); atomicAdd(p + 1, v.y);
  atomicAdd(p + 2, v.z); atomicAdd(p + 3, v.w);
}

__global__ void edge_agg1_k(const int* __restrict__ src, const int* __restrict__ dst,
                            const float* __restrict__ hsrc, float* __restrict__ pooled,
                            int D) {
  unsigned total = (unsigned)N_EDGES * (unsigned)D;
  unsigned t = blockIdx.x * blockDim.x + threadIdx.x;
  if (t >= total) return;
  unsigned e = t / D, c = t % D;
  atomicAdd(pooled + (size_t)dst[e] * D + c, hsrc[(size_t)src[e] * D + c]);
}

__global__ void graph_pool4_k(const float* __restrict__ h, const int* __restrict__ gid,
                              float* __restrict__ gp, int D) {
  unsigned D4 = (unsigned)D >> 2;
  unsigned total = (unsigned)N_NODES * D4;
  unsigned t = blockIdx.x * blockDim.x + threadIdx.x;
  if (t >= total) return;
  unsigned nidx = t / D4;
  unsigned c = (t % D4) << 2;
  int g = gid[nidx];
  float4 v = *(const float4*)(h + (size_t)nidx * D + c);
  float* p = gp + (size_t)g * D + c;
  atomicAdd(p + 0, v.x); atomicAdd(p + 1, v.y);
  atomicAdd(p + 2, v.z); atomicAdd(p + 3, v.w);
}

// first hyper layer: X is N x 2, W0 is 2 x HID
__global__ void hyper_first_k(float* __restrict__ Y, const float* __restrict__ p2,
                              const float* __restrict__ W0, const float* __restrict__ b0) {
  unsigned total = (unsigned)N_NODES * HID;
  unsigned t = blockIdx.x * blockDim.x + threadIdx.x;
  if (t >= total) return;
  unsigned nidx = t / HID, m = t % HID;
  Y[t] = p2[nidx * 2] * W0[m] + p2[nidx * 2 + 1] * W0[HID + m] + b0[m];
}

__global__ __launch_bounds__(256) void col_stats_k(const float* __restrict__ Y,
                                                   float* __restrict__ mean,
                                                   float* __restrict__ var,
                                                   int n, int M) {
  __shared__ float sh[256], sh2[256];
  int c = blockIdx.x;
  float s = 0.0f, s2 = 0.0f;
  for (int r = threadIdx.x; r < n; r += blockDim.x) {
    float v = Y[(size_t)r * M + c];
    s += v; s2 += v * v;
  }
  sh[threadIdx.x] = s; sh2[threadIdx.x] = s2;
  __syncthreads();
  for (int o = 128; o > 0; o >>= 1) {
    if (threadIdx.x < o) {
      sh[threadIdx.x]  += sh[threadIdx.x + o];
      sh2[threadIdx.x] += sh2[threadIdx.x + o];
    }
    __syncthreads();
  }
  if (threadIdx.x == 0) {
    float m = sh[0] / (float)n;
    mean[c] = m;
    var[c]  = sh2[0] / (float)n - m * m;
  }
}

__global__ void bn_relu_k(float* __restrict__ out, const float* __restrict__ y,
                          const float* __restrict__ mean, const float* __restrict__ var,
                          const float* __restrict__ g, const float* __restrict__ b,
                          unsigned total, int M) {
  unsigned t = blockIdx.x * blockDim.x + threadIdx.x;
  if (t >= total) return;
  int c = t % M;
  float v = g[c] * (y[t] - mean[c]) * rsqrtf(var[c] + BN_EPS) + b[c];
  out[t] = v > 0.0f ? v : 0.0f;
}

__global__ void combine_k(float* __restrict__ hm, const float* __restrict__ hh,
                          const float* __restrict__ atts, unsigned total) {
  unsigned t = blockIdx.x * blockDim.x + threadIdx.x;
  if (t >= total) return;
  float a0 = atts[0], a1 = atts[1];
  float mx = fmaxf(a0, a1);
  float e0 = __expf(a0 - mx), e1 = __expf(a1 - mx);
  float inv = 1.0f / (e0 + e1);
  hm[t] = (e0 * inv) * hm[t] + (e1 * inv) * hh[t];
}

__global__ void readout_k(float* __restrict__ score, const float* __restrict__ gp,
                          const float* __restrict__ W, const float* __restrict__ b,
                          int D, int init) {
  int t = blockIdx.x * blockDim.x + threadIdx.x;
  if (t >= N_G * DOUT) return;
  int g = t / DOUT, o = t % DOUT;
  float s = b[o];
  const float* row = gp + (size_t)g * D;
  for (int d = 0; d < D; ++d) s += row[d] * W[d * DOUT + o];
  if (init) score[t] = s;
  else      score[t] += s;
}

// ------------------------------ host side ----------------------------------

static inline int cdiv(long long a, int b) { return (int)((a + b - 1) / b); }

static void run_gemm(float* Y, const float* X, const float* W, const float* bias,
                     unsigned short* wbf, int K, int M, hipStream_t s) {
  convert_wt_k<<<cdiv(K * M, 256), 256, 0, s>>>(wbf, W, K, M);
  dim3 g(cdiv(N_NODES, BM), M / BN);
  gemm_bf16_k<<<g, 256, 0, s>>>(Y, X, wbf, bias, N_NODES, K, M);
}

static void run_bn_relu(float* out, const float* y, const float* g, const float* b,
                        float* statm, float* statv, int M, hipStream_t s) {
  col_stats_k<<<M, 256, 0, s>>>(y, statm, statv, N_NODES, M);
  unsigned tot = (unsigned)N_NODES * M;
  bn_relu_k<<<cdiv(tot, 256), 256, 0, s>>>(out, y, statm, statv, g, b, tot, M);
}

extern "C" void kernel_launch(void* const* d_in, const int* in_sizes, int n_in,
                              void* d_out, int out_size, void* d_ws, size_t ws_size,
                              hipStream_t stream) {
  // inputs (setup_inputs order; params tree-flattened with sorted dict keys)
  const float* x    = (const float*)d_in[0];
  const float* xh   = (const float*)d_in[1];
  const int*   esrc = (const int*)d_in[2];
  const int*   edst = (const int*)d_in[3];
  const int*   gid  = (const int*)d_in[4];
  const float* atts = (const float*)d_in[5];
  const float* bn_b[2]  = {(const float*)d_in[6],  (const float*)d_in[8]};
  const float* bn_g[2]  = {(const float*)d_in[7],  (const float*)d_in[9]};
  const float* eps      =  (const float*)d_in[10];
  const float* hbn_b[2] = {(const float*)d_in[11], (const float*)d_in[13]};
  const float* hbn_g[2] = {(const float*)d_in[12], (const float*)d_in[14]};
  // hyper_mlps[l]: W0,W1,b0,b1,be0,g0 at 15+6l ; mlps[l] at 27+6l ; pred[l] at 39+2l
  const float *hW0[2], *hW1[2], *hb0[2], *hb1[2], *hbe0[2], *hg0[2];
  const float *mW0[2], *mW1[2], *mb0[2], *mb1[2], *mbe0[2], *mg0[2];
  for (int l = 0; l < 2; ++l) {
    int base = 15 + 6 * l;
    hW0[l]  = (const float*)d_in[base + 0];
    hW1[l]  = (const float*)d_in[base + 1];
    hb0[l]  = (const float*)d_in[base + 2];
    hb1[l]  = (const float*)d_in[base + 3];
    hbe0[l] = (const float*)d_in[base + 4];
    hg0[l]  = (const float*)d_in[base + 5];
    base = 27 + 6 * l;
    mW0[l]  = (const float*)d_in[base + 0];
    mW1[l]  = (const float*)d_in[base + 1];
    mb0[l]  = (const float*)d_in[base + 2];
    mb1[l]  = (const float*)d_in[base + 3];
    mbe0[l] = (const float*)d_in[base + 4];
    mg0[l]  = (const float*)d_in[base + 5];
  }
  const float* predW[3] = {(const float*)d_in[39], (const float*)d_in[41], (const float*)d_in[43]};
  const float* predb[3] = {(const float*)d_in[40], (const float*)d_in[42], (const float*)d_in[44]};

  // workspace layout
  float* ws = (float*)d_ws;
  size_t NH = (size_t)N_NODES * HID;
  float* pooled = ws;                         // N x HID
  float* ybuf   = ws + NH;                    // N x HID
  float* h      = ws + 2 * NH;                // N x HID
  float* h2     = ws + 3 * NH;                // N x HID
  float* p2     = ws + 4 * NH;                // N x 2
  float* statm  = p2 + (size_t)N_NODES * 2;   // HID
  float* statv  = statm + HID;                // HID
  float* gp     = statv + HID;                // G x HID
  unsigned short* wbf = (unsigned short*)(gp + (size_t)N_G * HID);  // HID*HID bf16
  float* score = (float*)d_out;

  const unsigned nDin = (unsigned)N_NODES * DIM_IN;
  const unsigned nHid = (unsigned)N_NODES * HID;

  // ---- readout term for hidden[0] = x ----
  zero_k<<<cdiv(N_G * DIM_IN, 256), 256, 0, stream>>>(gp, N_G * DIM_IN);
  graph_pool4_k<<<cdiv((long long)N_NODES * (DIM_IN / 4), 256), 256, 0, stream>>>(x, gid, gp, DIM_IN);
  readout_k<<<cdiv(N_G * DOUT, 256), 256, 0, stream>>>(score, gp, predW[0], predb[0], DIM_IN, 1);

  // ---- layer 0: main stream ----
  scale_init_k<<<cdiv(nDin, 256), 256, 0, stream>>>(pooled, x, eps, 0, nDin);
  edge_agg4_k<<<cdiv((long long)N_EDGES * (DIM_IN / 4), 256), 256, 0, stream>>>(esrc, edst, x, pooled, DIM_IN);
  run_gemm(ybuf, pooled, mW0[0], mb0[0], wbf, DIM_IN, HID, stream);
  run_bn_relu(pooled, ybuf, mg0[0], mbe0[0], statm, statv, HID, stream);
  run_gemm(ybuf, pooled, mW1[0], mb1[0], wbf, HID, HID, stream);
  run_bn_relu(h, ybuf, bn_g[0], bn_b[0], statm, statv, HID, stream);

  // ---- layer 0: hyper stream ----
  scale_init_k<<<cdiv(N_NODES * 2, 256), 256, 0, stream>>>(p2, xh, eps, 0, (unsigned)N_NODES * 2);
  edge_agg1_k<<<cdiv((long long)N_EDGES * 2, 256), 256, 0, stream>>>(esrc, edst, xh, p2, 2);
  hyper_first_k<<<cdiv(nHid, 256), 256, 0, stream>>>(ybuf, p2, hW0[0], hb0[0]);
  run_bn_relu(pooled, ybuf, hg0[0], hbe0[0], statm, statv, HID, stream);
  run_gemm(ybuf, pooled, hW1[0], hb1[0], wbf, HID, HID, stream);
  run_bn_relu(h2, ybuf, hbn_g[0], hbn_b[0], statm, statv, HID, stream);

  combine_k<<<cdiv(nHid, 256), 256, 0, stream>>>(h, h2, atts, nHid);

  // ---- readout term for hidden[1] ----
  zero_k<<<cdiv(N_G * HID, 256), 256, 0, stream>>>(gp, N_G * HID);
  graph_pool4_k<<<cdiv((long long)N_NODES * (HID / 4), 256), 256, 0, stream>>>(h, gid, gp, HID);
  readout_k<<<cdiv(N_G * DOUT, 256), 256, 0, stream>>>(score, gp, predW[1], predb[1], HID, 0);

  // ---- layer 1: main stream ----
  scale_init_k<<<cdiv(nHid, 256), 256, 0, stream>>>(pooled, h, eps, 1, nHid);
  edge_agg4_k<<<cdiv((long long)N_EDGES * (HID / 4), 256), 256, 0, stream>>>(esrc, edst, h, pooled, HID);
  run_gemm(ybuf, pooled, mW0[1], mb0[1], wbf, HID, HID, stream);
  run_bn_relu(pooled, ybuf, mg0[1], mbe0[1], statm, statv, HID, stream);
  run_gemm(ybuf, pooled, mW1[1], mb1[1], wbf, HID, HID, stream);
  run_bn_relu(h, ybuf, bn_g[1], bn_b[1], statm, statv, HID, stream);

  // ---- layer 1: hyper stream ----
  scale_init_k<<<cdiv(nHid, 256), 256, 0, stream>>>(pooled, h2, eps, 1, nHid);
  edge_agg4_k<<<cdiv((long long)N_EDGES * (HID / 4), 256), 256, 0, stream>>>(esrc, edst, h2, pooled, HID);
  run_gemm(ybuf, pooled, hW0[1], hb0[1], wbf, HID, HID, stream);
  run_bn_relu(pooled, ybuf, hg0[1], hbe0[1], statm, statv, HID, stream);
  run_gemm(ybuf, pooled, hW1[1], hb1[1], wbf, HID, HID, stream);
  run_bn_relu(h2, ybuf, hbn_g[1], hbn_b[1], statm, statv, HID, stream);

  combine_k<<<cdiv(nHid, 256), 256, 0, stream>>>(h, h2, atts, nHid);

  // ---- readout term for hidden[2] ----
  zero_k<<<cdiv(N_G * HID, 256), 256, 0, stream>>>(gp, N_G * HID);
  graph_pool4_k<<<cdiv((long long)N_NODES * (HID / 4), 256), 256, 0, stream>>>(h, gid, gp, HID);
  readout_k<<<cdiv(N_G * DOUT, 256), 256, 0, stream>>>(score, gp, predW[2], predb[2], HID, 0);
}
